// Encoder_48412871360843
// MI455X (gfx1250) — matrix operand order repaired and checked by reading, not compiled, not run on hardware
//
#include <hip/hip_runtime.h>

// ---------------- problem constants ----------------
constexpr int B_   = 2;
constexpr int L_   = 2048;
constexpr int D_   = 1024;
constexpr int H_   = 16;
constexpr int DK_  = 64;
constexpr int DFF_ = 4096;
constexpr float EPS_ = 1e-5f;

// exp(s * 1/sqrt(64)) == exp2(s * (0.125 * log2(e)))  -- fold constants
constexpr float SCALE_LOG2E = 0.125f * 1.4426950408889634f;

typedef __attribute__((ext_vector_type(16))) __bf16          v16bf;
typedef __attribute__((ext_vector_type(8)))  float           v8f;
typedef __attribute__((ext_vector_type(16))) unsigned short  v16u;

__device__ __forceinline__ unsigned short f32_to_bf16_fast(float f) {
    return (unsigned short)((__builtin_bit_cast(unsigned, f) + 0x8000u) >> 16);
}
// pack two floats as bf16 pair in one v_perm_b32
__device__ __forceinline__ unsigned int pack_bf16x2(float a, float b) {
    unsigned ua = __builtin_bit_cast(unsigned, a) + 0x8000u;
    unsigned ub = __builtin_bit_cast(unsigned, b) + 0x8000u;
    return __builtin_amdgcn_perm(ua, ub, 0x03020706);   // {bf16(b),bf16(a)}
}
// raw v_exp_f32 (args are bounded; no denormal-range guard needed)
__device__ __forceinline__ float fast_exp2(float x) {
    return __builtin_amdgcn_exp2f(x);
}

// Fragment fetch from swizzled LDS: each lane's 16 bf16 elements are contiguous
// (32 bytes, 32B-aligned) -> 2x ds_load_b128, no repacking VALU.
__device__ __forceinline__ v16bf frag_lds(const unsigned short* swz, int tile, int lane) {
    return __builtin_bit_cast(v16bf,
        *(const v16u*)(swz + (((size_t)(tile * 32 + lane)) << 4)));
}

// ---------------------------------------------------------------------------
// LayerNorm over joint (L,D) per batch: partial reduce -> final -> apply
// ---------------------------------------------------------------------------
constexpr int LN_PB = 256;   // partial blocks per batch

__global__ void __launch_bounds__(256)
ln_partial_kernel(const float* __restrict__ x, float* __restrict__ partials) {
    constexpr size_t LD = (size_t)L_ * D_;
    const int b   = blockIdx.x / LN_PB;
    const int blk = blockIdx.x % LN_PB;
    const size_t chunk = LD / LN_PB;                 // 8192
    const size_t base  = (size_t)b * LD + (size_t)blk * chunk;
    const float4* x4 = (const float4*)(x + base);
    float s = 0.f, s2 = 0.f;
    for (size_t i = threadIdx.x; i < chunk / 4; i += 256) {
        float4 v = x4[i];
        s  += v.x + v.y + v.z + v.w;
        s2 += v.x*v.x + v.y*v.y + v.z*v.z + v.w*v.w;
    }
    __shared__ float sh[512];
    sh[threadIdx.x] = s; sh[256 + threadIdx.x] = s2;
    __syncthreads();
    for (int ofs = 128; ofs > 0; ofs >>= 1) {
        if ((int)threadIdx.x < ofs) {
            sh[threadIdx.x]       += sh[threadIdx.x + ofs];
            sh[256 + threadIdx.x] += sh[256 + threadIdx.x + ofs];
        }
        __syncthreads();
    }
    if (threadIdx.x == 0) {
        partials[2 * blockIdx.x]     = sh[0];
        partials[2 * blockIdx.x + 1] = sh[256];
    }
}

__global__ void __launch_bounds__(256)
ln_final_kernel(const float* __restrict__ partials, float* __restrict__ stats) {
    constexpr size_t LD = (size_t)L_ * D_;
    const int b = blockIdx.x;
    float s  = partials[2 * (b * LN_PB + threadIdx.x)];
    float s2 = partials[2 * (b * LN_PB + threadIdx.x) + 1];
    __shared__ float sh[512];
    sh[threadIdx.x] = s; sh[256 + threadIdx.x] = s2;
    __syncthreads();
    for (int ofs = 128; ofs > 0; ofs >>= 1) {
        if ((int)threadIdx.x < ofs) {
            sh[threadIdx.x]       += sh[threadIdx.x + ofs];
            sh[256 + threadIdx.x] += sh[256 + threadIdx.x + ofs];
        }
        __syncthreads();
    }
    if (threadIdx.x == 0) {
        float mean = sh[0] / (float)LD;
        float var  = sh[256] / (float)LD - mean * mean;
        stats[2 * b]     = mean;
        stats[2 * b + 1] = rsqrtf(var + EPS_);
    }
}

__global__ void __launch_bounds__(256)
ln_apply_kernel(const float* __restrict__ x, const float* __restrict__ w,
                const float* __restrict__ bb, const float* __restrict__ stats,
                float* __restrict__ out) {
    constexpr size_t LD    = (size_t)L_ * D_;
    constexpr size_t total = (size_t)B_ * LD;
    size_t i = (size_t)blockIdx.x * 256 + threadIdx.x;
    if (i >= total) return;
    int b = (int)(i / LD);
    size_t r = i - (size_t)b * LD;
    float mean = stats[2 * b], rstd = stats[2 * b + 1];
    out[i] = (x[i] - mean) * rstd * w[r] + bb[r];
}

// ---------------------------------------------------------------------------
// Tiled GEMM: out[M,N] = act( A[M,K] @ W[K,N] + bias ) (+ res)
// 128x128x64 stage (two 32-k chunks per barrier round), 8 waves (2x4),
// wave tile 64x32 -> 4x2 WMMA accumulators, 16 WMMAs per stage.
// LDS staged pre-swizzled into the exact WMMA fragment layout.
// ---------------------------------------------------------------------------
template<bool A_BF16, bool RELU, bool HAS_RES, bool OUT_BF16>
__global__ void __launch_bounds__(256)
gemm_wmma_kernel(const void* __restrict__ A, const float* __restrict__ W,
                 const float* __restrict__ bias, const float* __restrict__ res,
                 void* __restrict__ out, int M, int N, int K) {
    constexpr int BM = 128, BN = 128, BK = 64;
    constexpr int CH = 8 * 32 * 16;                  // ushorts per 32-k chunk
    __shared__ __align__(32) unsigned short swzA[2 * CH];   // 2 chunks x 8 M-tiles
    __shared__ __align__(32) unsigned short swzB[2 * CH];   // 2 chunks x 8 N-tiles

    const int m0 = blockIdx.y * BM;
    const int n0 = blockIdx.x * BN;
    const int tid  = threadIdx.x;
    const int wave = tid >> 5;
    const int lane = tid & 31;
    const int wm = wave >> 2;        // 0..1
    const int wn = wave & 3;         // 0..3

    const v8f vzero = {0.f, 0.f, 0.f, 0.f, 0.f, 0.f, 0.f, 0.f};
    v8f acc[4][2];
#pragma unroll
    for (int i = 0; i < 4; ++i)
#pragma unroll
        for (int j = 0; j < 2; ++j) acc[i][j] = vzero;

    for (int k0 = 0; k0 < K; k0 += BK) {
        __syncthreads();
        // --- stage A tile: quads of 4 k-values (stay inside one frag half) ---
        if (A_BF16) {
            const unsigned int* Ah = (const unsigned int*)A;
            for (int e4 = tid; e4 < BM * BK / 4; e4 += 256) {
                const int m = e4 >> 4, kk = (e4 & 15) * 4;
                uint2 wv = *(const uint2*)&Ah[((size_t)(m0 + m) * K + k0 + kk) >> 1];
                const int chunk = kk >> 5, k = kk & 31;
                const int e0 = (k & 7) + ((k >= 16) ? 8 : 0);
                *(uint2*)&swzA[chunk * CH +
                    ((((m >> 4) * 32 + (m & 15) + ((k >> 3) & 1) * 16)) << 4) + e0] = wv;
            }
        } else {
            const float* Af = (const float*)A;
            for (int e4 = tid; e4 < BM * BK / 4; e4 += 256) {
                const int m = e4 >> 4, kk = (e4 & 15) * 4;
                float4 v = *(const float4*)&Af[(size_t)(m0 + m) * K + k0 + kk];
                uint2 wv = { pack_bf16x2(v.x, v.y), pack_bf16x2(v.z, v.w) };
                const int chunk = kk >> 5, k = kk & 31;
                const int e0 = (k & 7) + ((k >= 16) ? 8 : 0);
                *(uint2*)&swzA[chunk * CH +
                    ((((m >> 4) * 32 + (m & 15) + ((k >> 3) & 1) * 16)) << 4) + e0] = wv;
            }
        }
        // --- stage B tile (f32 weights -> bf16) into fragment slots ---
        for (int e2 = tid; e2 < BK * BN / 2; e2 += 256) {
            const int n = e2 & 127, kk = (e2 >> 7) * 2;
            float b0 = W[(size_t)(k0 + kk)     * N + n0 + n];
            float b1 = W[(size_t)(k0 + kk + 1) * N + n0 + n];
            unsigned int wv = pack_bf16x2(b0, b1);
            const int chunk = kk >> 5, k = kk & 31;
            ((unsigned int*)swzB)[(chunk * 8 + (n >> 4)) * 256 +
                                  ((n & 15) + ((k >> 3) & 1) * 16) * 8 + ((k & 7) >> 1) + ((k >= 16) ? 4 : 0)] = wv;
        }
        if (k0 + BK < K) {  // hint next W tile toward L2 (global_prefetch_b8)
            __builtin_prefetch(&W[(size_t)(k0 + BK) * N + n0 + (tid & 127)], 0, 1);
        }
        __syncthreads();

#pragma unroll
        for (int kc = 0; kc < 2; ++kc) {
            v16bf a[4], b[2];
#pragma unroll
            for (int i = 0; i < 4; ++i) a[i] = frag_lds(swzA + kc * CH, wm * 4 + i, lane);
#pragma unroll
            for (int j = 0; j < 2; ++j) b[j] = frag_lds(swzB + kc * CH, wn * 2 + j, lane);
#pragma unroll
            for (int i = 0; i < 4; ++i)
#pragma unroll
                for (int j = 0; j < 2; ++j)
                    acc[i][j] = __builtin_amdgcn_wmma_f32_16x16x32_bf16(
                        false, a[i], false, b[j], (short)0, acc[i][j], false, false);
        }
    }

    // --- epilogue: bias (+relu) (+res), store ---
    const int hi = (lane >> 4) & 1;
    const int ln = lane & 15;
#pragma unroll
    for (int i = 0; i < 4; ++i) {
#pragma unroll
        for (int j = 0; j < 2; ++j) {
            const int col = n0 + wn * 32 + j * 16 + ln;
            const float bvl = bias[col];
#pragma unroll
            for (int r = 0; r < 8; ++r) {
                const int row = m0 + wm * 64 + i * 16 + r + hi * 8;
                float v = acc[i][j][r] + bvl;
                if (RELU) v = fmaxf(v, 0.f);
                const size_t idx = (size_t)row * N + col;
                if (HAS_RES) v += res[idx];
                if (OUT_BF16) ((unsigned short*)out)[idx] = f32_to_bf16_fast(v);
                else          ((float*)out)[idx] = v;
            }
        }
    }
}

// ---------------------------------------------------------------------------
// Attention, tril-zero semantics: masked scores are ZERO (not -inf) and still
// contribute exp(0)=1. Scores are bounded, so fixed-shift softmax is exact:
//   p = exp2(s*0.125*log2e) (masked -> 1), out = (P @ V) / (P @ ones).
// Row sums via WMMA against a ones fragment (no shuffles). Fully-masked tiles
// use ones@V and an analytic l contribution. Wave-uniform branch conditions go
// through readfirstlane so they compile to scalar branches (EXEC untouched,
// as WMMA requires).
// ---------------------------------------------------------------------------
constexpr int ATT_WAVES = 4;

__global__ void __launch_bounds__(32 * ATT_WAVES)
attention_kernel(const unsigned short* __restrict__ Q,
                 const unsigned short* __restrict__ Km,
                 const unsigned short* __restrict__ V,
                 unsigned short* __restrict__ ctx) {
    __shared__ __align__(32) unsigned short Ks[4 * 32 * 16];           // 4 S B-frag tiles
    __shared__ __align__(32) unsigned short Vs[4 * 32 * 16];           // 4 PV B-frag tiles
    __shared__ __align__(32) unsigned short pbuf[ATT_WAVES * 32 * 16]; // P A-frag per wave

    const int wave = threadIdx.x >> 5;
    const int lane = threadIdx.x & 31;
    const int hi   = (lane >> 4) & 1;
    const int ln   = lane & 15;
    const int b = blockIdx.z, h = blockIdx.y;
    const int q0 = (blockIdx.x * ATT_WAVES + wave) * 16;
    const size_t rowbase = (size_t)b * L_;
    const int dbase = h * DK_;
    const v8f vzero = {0.f, 0.f, 0.f, 0.f, 0.f, 0.f, 0.f, 0.f};

    // ones fragment (bf16 1.0 = 0x3F80): A 16x32 of ones / B 32x16 of ones
    v16bf ones16;
    {
        v16u uo;
#pragma unroll
        for (int i = 0; i < 16; ++i) uo[i] = 0x3F80;
        ones16 = __builtin_bit_cast(v16bf, uo);
    }

    // wave-uniform tile classification, forced into SGPRs for scalar branches
    const int q0s         = __builtin_amdgcn_readfirstlane(q0);
    const int first_above = __builtin_amdgcn_readfirstlane((q0 + 16 + 31) & ~31);
    const float lextra = (float)(L_ - first_above);   // exp(0)=1 terms, analytic
    const int k_need_end = blockIdx.x * (ATT_WAVES * 16) + 64;

    // Q fragments: 16x64 -> two 16x32 A-frags (one-time global gather)
    v16bf qa[2];
    {
        const unsigned short* qrow = Q + (rowbase + q0 + ln) * D_ + dbase;
#pragma unroll
        for (int c = 0; c < 2; ++c) {
            v16u u;
#pragma unroll
            for (int i = 0; i < 8; ++i) {
                int k0 = 2 * i + ((i < 4) ? 0 : 8) + (hi ? 8 : 0);
                u[2 * i]     = qrow[c * 32 + k0];
                u[2 * i + 1] = qrow[c * 32 + k0 + 1];
            }
            qa[c] = __builtin_bit_cast(v16bf, u);
        }
    }

    v8f accv[4], lacc = vzero;
#pragma unroll
    for (int t = 0; t < 4; ++t) accv[t] = vzero;

    unsigned short* pw_base = pbuf + wave * 512;

    for (int j0 = 0; j0 < L_; j0 += 32) {
        __syncthreads();
        // ---- cooperative staging of K (while needed) and V tiles ----
        const bool need_k = (j0 < k_need_end);
        for (int q4 = (int)threadIdx.x; q4 < 512; q4 += 32 * ATT_WAVES) {
            const int key = q4 >> 4;            // 0..31
            const int dd  = (q4 & 15) * 4;      // 0..60
            const size_t gro = (rowbase + j0 + key) * D_ + dbase + dd;
            if (need_k) {
                uint2 wk = *(const uint2*)(Km + gro);
                const int c = dd >> 5, dk = dd & 31;
                const int kslot = ((((key >> 4) * 2 + c) * 32 + (key & 15) +
                                    ((dk >> 3) & 1) * 16) << 4) + (dk & 7) + ((dk >= 16) ? 8 : 0);
                *(uint2*)&Ks[kslot] = wk;
            }
            uint2 wv = *(const uint2*)(V + gro);
            unsigned short vv[4] = {
                (unsigned short)(wv.x & 0xffff), (unsigned short)(wv.x >> 16),
                (unsigned short)(wv.y & 0xffff), (unsigned short)(wv.y >> 16) };
            const int dt = dd >> 4, vhi = (key >> 3) & 1, ve = (key & 7) + ((key >> 4) & 1) * 8;
            const int dlo = dd & 15;
#pragma unroll
            for (int jj = 0; jj < 4; ++jj)
                Vs[((dt * 32 + dlo + jj + vhi * 16) << 4) + ve] = vv[jj];
        }
        if (j0 + 32 < L_) {
            __builtin_prefetch(V + (rowbase + j0 + 32 + (threadIdx.x & 31)) * D_ + dbase, 0, 1);
            if (j0 + 32 < k_need_end)
                __builtin_prefetch(Km + (rowbase + j0 + 32 + (threadIdx.x & 31)) * D_ + dbase, 0, 1);
        }
        __syncthreads();

        if (j0 >= first_above) {
            // ---- fully masked tile: P == 1. l handled analytically. ----
#pragma unroll
            for (int dt = 0; dt < 4; ++dt)
                accv[dt] = __builtin_amdgcn_wmma_f32_16x16x32_bf16(
                    false, ones16, false, frag_lds(Vs, dt, lane),
                    (short)0, accv[dt], false, false);
        } else {
            // ----- S = Q K^T for this 16x32 key tile -----
            v8f s[2] = {vzero, vzero};
#pragma unroll
            for (int jt = 0; jt < 2; ++jt)
#pragma unroll
                for (int c = 0; c < 2; ++c)
                    s[jt] = __builtin_amdgcn_wmma_f32_16x16x32_bf16(
                        false, qa[c], false, frag_lds(Ks, jt * 2 + c, lane),
                        (short)0, s[jt], false, false);

            if (j0 + 31 <= q0s) {
                // fully below diagonal: no masks needed
#pragma unroll
                for (int r = 0; r < 8; ++r) {
                    const float p0 = fast_exp2(s[0][r] * SCALE_LOG2E);
                    const float p1 = fast_exp2(s[1][r] * SCALE_LOG2E);
                    unsigned short* pw = pw_base +
                        (((r + hi * 8) + ((ln >> 3) & 1) * 16) << 4) + (ln & 7);
                    pw[0] = f32_to_bf16_fast(p0);
                    pw[8] = f32_to_bf16_fast(p1);
                }
            } else {
                // boundary tile: per-element tril mask (masked -> exp(0)=1)
#pragma unroll
                for (int r = 0; r < 8; ++r) {
                    const int qi = q0 + r + hi * 8;
                    const float p0 = (j0 + ln      <= qi) ? fast_exp2(s[0][r] * SCALE_LOG2E) : 1.f;
                    const float p1 = (j0 + 16 + ln <= qi) ? fast_exp2(s[1][r] * SCALE_LOG2E) : 1.f;
                    unsigned short* pw = pw_base +
                        (((r + hi * 8) + ((ln >> 3) & 1) * 16) << 4) + (ln & 7);
                    pw[0] = f32_to_bf16_fast(p0);
                    pw[8] = f32_to_bf16_fast(p1);
                }
            }
            // ----- acc += P @ V; l += P @ ones (row sums via WMMA) -----
            v16bf pa = frag_lds(pw_base, 0, lane);
#pragma unroll
            for (int dt = 0; dt < 4; ++dt)
                accv[dt] = __builtin_amdgcn_wmma_f32_16x16x32_bf16(
                    false, pa, false, frag_lds(Vs, dt, lane),
                    (short)0, accv[dt], false, false);
            lacc = __builtin_amdgcn_wmma_f32_16x16x32_bf16(
                false, pa, false, ones16, (short)0, lacc, false, false);
        }
    }

    // ----- normalize and store ctx (bf16) -----
#pragma unroll
    for (int r = 0; r < 8; ++r) {
        const int m = q0 + r + hi * 8;
        const float linv = 1.0f / (lacc[r] + lextra);
#pragma unroll
        for (int dt = 0; dt < 4; ++dt) {
            const int d = dbase + dt * 16 + ln;
            ctx[(rowbase + m) * D_ + d] = f32_to_bf16_fast(accv[dt][r] * linv);
        }
    }
}

// ---------------------------------------------------------------------------
extern "C" void kernel_launch(void* const* d_in, const int* in_sizes, int n_in,
                              void* d_out, int out_size, void* d_ws, size_t ws_size,
                              hipStream_t stream) {
    (void)in_sizes; (void)n_in; (void)out_size; (void)ws_size;
    const float* x     = (const float*)d_in[0];
    const float* y     = (const float*)d_in[1];
    const float* Wq    = (const float*)d_in[2];
    const float* bq    = (const float*)d_in[3];
    const float* Wk    = (const float*)d_in[4];
    const float* bk    = (const float*)d_in[5];
    const float* Wv    = (const float*)d_in[6];
    const float* bv    = (const float*)d_in[7];
    const float* Wo    = (const float*)d_in[8];
    const float* bo    = (const float*)d_in[9];
    const float* W1    = (const float*)d_in[10];
    const float* b1    = (const float*)d_in[11];
    const float* W2    = (const float*)d_in[12];
    const float* b2    = (const float*)d_in[13];
    const float* ln1_w = (const float*)d_in[14];
    const float* ln1_b = (const float*)d_in[15];
    const float* ln2_w = (const float*)d_in[16];
    const float* ln2_b = (const float*)d_in[17];
    float* out = (float*)d_out;                       // y1 lives here too

    constexpr size_t LD  = (size_t)L_ * D_;           // 2,097,152
    constexpr size_t BLD = (size_t)B_ * LD;           // 4,194,304

    // workspace carve-up (~80 MB)
    char* ws = (char*)d_ws;
    float*          lny = (float*)ws;                 ws += BLD * sizeof(float);   // reused for ln2
    unsigned short* Qb  = (unsigned short*)ws;        ws += BLD * sizeof(unsigned short);
    unsigned short* Kb  = (unsigned short*)ws;        ws += BLD * sizeof(unsigned short);
    unsigned short* Vb  = (unsigned short*)ws;        ws += BLD * sizeof(unsigned short);
    unsigned short* ctx = (unsigned short*)ws;        ws += BLD * sizeof(unsigned short);
    unsigned short* hb  = (unsigned short*)ws;        ws += (size_t)BLD * 4 * sizeof(unsigned short);
    float* partials = (float*)ws;                     ws += (size_t)B_ * LN_PB * 2 * sizeof(float);
    float* stats    = (float*)ws;

    const int M = B_ * L_;                            // 4096
    const dim3 blk256(256);
    const dim3 gN1024(D_   / 128, M / 128);           // 8 x 32
    const dim3 gN4096(DFF_ / 128, M / 128);           // 32 x 32

    // ---- sublayer 0: LN(y) ----
    ln_partial_kernel<<<B_ * LN_PB, blk256, 0, stream>>>(y, partials);
    ln_final_kernel<<<B_, blk256, 0, stream>>>(partials, stats);
    ln_apply_kernel<<<(unsigned)((BLD + 255) / 256), blk256, 0, stream>>>(y, ln1_w, ln1_b, stats, lny);

    // ---- QKV projections (bf16 WMMA, f32 accumulate) ----
    gemm_wmma_kernel<false, false, false, true><<<gN1024, blk256, 0, stream>>>(
        (const void*)x,   Wq, bq, nullptr, (void*)Qb, M, D_, D_);
    gemm_wmma_kernel<false, false, false, true><<<gN1024, blk256, 0, stream>>>(
        (const void*)lny, Wk, bk, nullptr, (void*)Kb, M, D_, D_);
    gemm_wmma_kernel<false, false, false, true><<<gN1024, blk256, 0, stream>>>(
        (const void*)lny, Wv, bv, nullptr, (void*)Vb, M, D_, D_);

    // ---- attention (tril-zero semantics, full key scan) ----
    attention_kernel<<<dim3(L_ / (16 * ATT_WAVES), H_, B_), 32 * ATT_WAVES, 0, stream>>>(
        Qb, Kb, Vb, ctx);

    // ---- y1 = y + ctx @ Wo + bo  (into d_out) ----
    gemm_wmma_kernel<true, false, true, false><<<gN1024, blk256, 0, stream>>>(
        (const void*)ctx, Wo, bo, y, (void*)out, M, D_, D_);

    // ---- sublayer 1: LN(y1), FFN, residual ----
    ln_partial_kernel<<<B_ * LN_PB, blk256, 0, stream>>>(out, partials);
    ln_final_kernel<<<B_, blk256, 0, stream>>>(partials, stats);
    ln_apply_kernel<<<(unsigned)((BLD + 255) / 256), blk256, 0, stream>>>(out, ln2_w, ln2_b, stats, lny);

    gemm_wmma_kernel<false, true, false, true><<<gN4096, blk256, 0, stream>>>(
        (const void*)lny, W1, b1, nullptr, (void*)hb, M, DFF_, D_);
    gemm_wmma_kernel<true, false, true, false><<<gN1024, blk256, 0, stream>>>(
        (const void*)hb, W2, b2, out, (void*)out, M, D_, DFF_);
}
